// HybridGATGCN_10093173145790
// MI455X (gfx1250) — matrix operand order repaired and checked by reading, not compiled, not run on hardware
//
#include <hip/hip_runtime.h>
#include <hip/hip_bf16.h>
#include <math.h>

#define N_NODES   50000
#define NEG_SLOPE 0.2f
#define BN_EPS    1e-5f

typedef __attribute__((ext_vector_type(16))) _Float16 v16h;
typedef __attribute__((ext_vector_type(8)))  float    v8f;

union Frag16 { v16h v; unsigned int u[8]; };
union Frag32 { v8f  v; float        f[8]; };

// ---------------- elementwise helpers ----------------
__global__ void k_fill_f32(float* p, float val, long n) {
  long i = (long)blockIdx.x * blockDim.x + threadIdx.x;
  if (i < n) p[i] = val;
}

__global__ void k_zero_f32(float* p, long n) {
  long i = (long)blockIdx.x * blockDim.x + threadIdx.x;
  if (i < n) p[i] = 0.0f;
}

__global__ void k_cvt_f16(const float* __restrict__ in, _Float16* __restrict__ out, long n) {
  long i = (long)blockIdx.x * blockDim.x + threadIdx.x;
  if (i < n) out[i] = (_Float16)in[i];
}

// W [K,M] row-major -> Wt [M,K] f16 (so B-fragment K-pairs are contiguous)
__global__ void k_cvt_wT(const float* __restrict__ W, _Float16* __restrict__ Wt, int K, int M) {
  int i = blockIdx.x * blockDim.x + threadIdx.x;
  if (i < K * M) {
    int k = i / M, m = i % M;
    Wt[(long)m * K + k] = (_Float16)W[i];
  }
}

__global__ void k_deg(const int* __restrict__ dst, float* __restrict__ deg, int E) {
  int i = blockIdx.x * blockDim.x + threadIdx.x;
  if (i < E) atomicAdd(&deg[dst[i]], 1.0f);
}

__global__ void k_rsqrt_inplace(float* p, long n) {
  long i = (long)blockIdx.x * blockDim.x + threadIdx.x;
  if (i < n) p[i] = rsqrtf(p[i]);
}

// ---------------- WMMA GEMM: C[N,M] = A[N,K] * W[K,M]  (A f16 row-major, W given as Wt[M,K] f16)
__global__ void k_gemm_wmma(const _Float16* __restrict__ A,
                            const _Float16* __restrict__ Bt,
                            float* __restrict__ C,
                            int n, int K, int M) {
  int lane = threadIdx.x & 31;
  int wid  = (int)(((long)blockIdx.x * blockDim.x + threadIdx.x) >> 5);
  int mt = M >> 4;
  int tiles = (n >> 4) * mt;
  if (wid >= tiles) return;
  int rt = wid / mt, ct = wid % mt;
  int m0 = rt << 4, n0 = ct << 4;
  int half = lane >> 4;          // lane group 0-15 vs 16-31
  int sub  = lane & 15;
  const _Float16* ap = A  + (long)(m0 + sub) * K;   // A row for this lane
  const _Float16* bp = Bt + (long)(n0 + sub) * K;   // B column for this lane

  Frag32 acc;
#pragma unroll
  for (int r = 0; r < 8; ++r) acc.f[r] = 0.0f;

  for (int k0 = 0; k0 < K; k0 += 32) {
    Frag16 a, b;
#pragma unroll
    for (int v = 0; v < 8; ++v) {
      // ISA 7.12.2: 16-bit A 16x32 layout
      int ka = k0 + 2 * v + (v >= 4 ? 8 : 0) + 8 * half;
      // 16-bit B 32x16 layout (same K order as A)
      int kb = k0 + 2 * v + (v >= 4 ? 8 : 0) + 8 * half;
      a.u[v] = *(const unsigned int*)(ap + ka);
      b.u[v] = *(const unsigned int*)(bp + kb);
    }
    acc.v = __builtin_amdgcn_wmma_f32_16x16x32_f16(
        /*neg_a=*/false, a.v, /*neg_b=*/false, b.v,
        /*c_mod=*/(short)0, acc.v, /*reuse_a=*/false, /*reuse_b=*/false);
  }

#pragma unroll
  for (int r = 0; r < 8; ++r) {
    int m = m0 + r + 8 * half;       // D layout: VGPR r -> M=r (lanes 0-15) / M=r+8 (16-31)
    C[(long)m * M + n0 + sub] = acc.f[r];
  }
}

// ---------------- GCN scatter: out[d] += h[s] * dinv[s]*dinv[d]  (edges + self loops)
__global__ void k_gcn_scatter(const float* __restrict__ h,
                              const int* __restrict__ src, const int* __restrict__ dst,
                              const float* __restrict__ dinv,
                              float* __restrict__ out, int E, int C) {
  int wid  = (int)(((long)blockIdx.x * blockDim.x + threadIdx.x) >> 5);
  int lane = threadIdx.x & 31;
  int total = E + N_NODES;
  if (wid >= total) return;
  int s, d;
  if (wid < E) { s = src[wid]; d = dst[wid]; } else { s = d = wid - E; }
  float nrm = dinv[s] * dinv[d];
  const float* hp = h + (long)s * C;
  float* op = out + (long)d * C;
  __builtin_prefetch(hp, 0, 0);   // global_prefetch_b8
  for (int c = lane; c < C; c += 32)
    atomicAdd(op + c, hp[c] * nrm);
}

// ---------------- GCN epilogue: bias + eval BN + ReLU -> f16
__global__ void k_gcn_post(const float* __restrict__ in, const float* __restrict__ gb,
                           const float* __restrict__ g, const float* __restrict__ bb,
                           const float* __restrict__ m, const float* __restrict__ v,
                           _Float16* __restrict__ outh, long n, int C) {
  long i = (long)blockIdx.x * blockDim.x + threadIdx.x;
  if (i < n) {
    int c = (int)(i % C);
    float val = in[i] + gb[c];
    val = (val - m[c]) * rsqrtf(v[c] + BN_EPS) * g[c] + bb[c];
    outh[i] = (_Float16)fmaxf(val, 0.0f);
  }
}

// ---------------- GAT attention ----------------
__device__ __forceinline__ unsigned f2ord(float f) {
  unsigned b = __float_as_uint(f);
  return (b & 0x80000000u) ? ~b : (b | 0x80000000u);
}
__device__ __forceinline__ float ord2f(unsigned u) {
  return __uint_as_float((u & 0x80000000u) ? (u & 0x7FFFFFFFu) : ~u);
}
__device__ __forceinline__ float leaky(float e) { return e < 0.0f ? e * NEG_SLOPE : e; }

__global__ void k_gat_logits(const float* __restrict__ hm,
                             const float* __restrict__ as, const float* __restrict__ ad,
                             float* __restrict__ als, float* __restrict__ ald,
                             int n, int H, int C) {
  int i = blockIdx.x * blockDim.x + threadIdx.x;   // over n*H
  if (i < n * H) {
    int node = i / H, h = i % H;
    const float* hp = hm + (long)node * H * C + (long)h * C;
    const float* ap = as + h * C;
    const float* dp = ad + h * C;
    float ss = 0.0f, sd = 0.0f;
    for (int c = 0; c < C; ++c) { float x = hp[c]; ss += x * ap[c]; sd += x * dp[c]; }
    als[i] = ss; ald[i] = sd;
  }
}

__global__ void k_att_max(const int* __restrict__ src, const int* __restrict__ dst,
                          const float* __restrict__ als, const float* __restrict__ ald,
                          unsigned* __restrict__ emax, int E, int H) {
  int i = blockIdx.x * blockDim.x + threadIdx.x;
  if (i >= E + N_NODES) return;
  int s, d;
  if (i < E) { s = src[i]; d = dst[i]; } else { s = d = i - E; }
  for (int h = 0; h < H; ++h) {
    float e = leaky(als[s * H + h] + ald[d * H + h]);
    atomicMax(emax + d * H + h, f2ord(e));
  }
}

__global__ void k_att_denom(const int* __restrict__ src, const int* __restrict__ dst,
                            const float* __restrict__ als, const float* __restrict__ ald,
                            const unsigned* __restrict__ emax, float* __restrict__ denom,
                            int E, int H) {
  int i = blockIdx.x * blockDim.x + threadIdx.x;
  if (i >= E + N_NODES) return;
  int s, d;
  if (i < E) { s = src[i]; d = dst[i]; } else { s = d = i - E; }
  for (int h = 0; h < H; ++h) {
    float e = leaky(als[s * H + h] + ald[d * H + h]);
    atomicAdd(denom + d * H + h, expf(e - ord2f(emax[d * H + h])));
  }
}

__global__ void k_gat_scatter(const float* __restrict__ hm,
                              const int* __restrict__ src, const int* __restrict__ dst,
                              const float* __restrict__ als, const float* __restrict__ ald,
                              const unsigned* __restrict__ emax, const float* __restrict__ denom,
                              float* __restrict__ out, int E, int H, int C) {
  int wid  = (int)(((long)blockIdx.x * blockDim.x + threadIdx.x) >> 5);
  int lane = threadIdx.x & 31;
  if (wid >= E + N_NODES) return;
  int s, d;
  if (wid < E) { s = src[wid]; d = dst[wid]; } else { s = d = wid - E; }
  float e0 = leaky(als[s * H + 0] + ald[d * H + 0]);
  float e1 = leaky(als[s * H + 1] + ald[d * H + 1]);
  float a0 = expf(e0 - ord2f(emax[d * H + 0])) / denom[d * H + 0];
  float a1 = expf(e1 - ord2f(emax[d * H + 1])) / denom[d * H + 1];
  int HC = H * C;
  const float* hp = hm + (long)s * HC;
  float* op = out + (long)d * HC;
  __builtin_prefetch(hp, 0, 0);
  for (int c = lane; c < HC; c += 32) {
    float al = (c < C) ? a0 : a1;
    atomicAdd(op + c, hp[c] * al);
  }
}

// head mean + bias + ReLU -> f16 (layer-3 input)
__global__ void k_gat_post_relu(const float* __restrict__ acc, const float* __restrict__ b,
                                _Float16* __restrict__ outh, long n, int C) {
  long i = (long)blockIdx.x * blockDim.x + threadIdx.x;
  if (i < n) {
    long node = i / C; int c = (int)(i % C);
    float v = 0.5f * (acc[node * 2 * C + c] + acc[node * 2 * C + C + c]) + b[c];
    outh[i] = (_Float16)fmaxf(v, 0.0f);
  }
}

// final: head mean + bias + log_softmax over C (wave per node, wave32 reductions)
__global__ void k_final_logsoftmax(const float* __restrict__ acc, const float* __restrict__ b,
                                   float* __restrict__ out, int n, int C) {
  int wid  = (int)(((long)blockIdx.x * blockDim.x + threadIdx.x) >> 5);
  int lane = threadIdx.x & 31;
  if (wid >= n) return;
  const float* ap = acc + (long)wid * 2 * C;
  int c0 = lane, c1 = lane + 32;
  float v0 = -INFINITY, v1 = -INFINITY;
  if (c0 < C) v0 = 0.5f * (ap[c0] + ap[C + c0]) + b[c0];
  if (c1 < C) v1 = 0.5f * (ap[c1] + ap[C + c1]) + b[c1];
  float mx = fmaxf(v0, v1);
  for (int off = 16; off; off >>= 1) mx = fmaxf(mx, __shfl_xor(mx, off, 32));
  float s = (c0 < C ? expf(v0 - mx) : 0.0f) + (c1 < C ? expf(v1 - mx) : 0.0f);
  for (int off = 16; off; off >>= 1) s += __shfl_xor(s, off, 32);
  float ls = logf(s);
  if (c0 < C) out[(long)wid * C + c0] = v0 - mx - ls;
  if (c1 < C) out[(long)wid * C + c1] = v1 - mx - ls;
}

// ---------------- host launcher ----------------
extern "C" void kernel_launch(void* const* d_in, const int* in_sizes, int n_in,
                              void* d_out, int out_size, void* d_ws, size_t ws_size,
                              hipStream_t stream) {
  (void)n_in; (void)out_size; (void)ws_size;
  const int N = N_NODES;
  const float* x      = (const float*)d_in[0];
  const int*   ei     = (const int*)d_in[1];
  const int    E      = in_sizes[1] / 2;
  const int*   src    = ei;
  const int*   dst    = ei + E;
  const float* gcn1_W = (const float*)d_in[2];
  const float* gcn1_b = (const float*)d_in[3];
  const float* bn1_g  = (const float*)d_in[4];
  const float* bn1_b  = (const float*)d_in[5];
  const float* bn1_m  = (const float*)d_in[6];
  const float* bn1_v  = (const float*)d_in[7];
  const float* gat1_W = (const float*)d_in[8];
  const float* gat1_as= (const float*)d_in[9];
  const float* gat1_ad= (const float*)d_in[10];
  const float* gat1_b = (const float*)d_in[11];
  const float* gcn2_W = (const float*)d_in[12];
  const float* gcn2_b = (const float*)d_in[13];
  const float* bn2_g  = (const float*)d_in[14];
  const float* bn2_b  = (const float*)d_in[15];
  const float* bn2_m  = (const float*)d_in[16];
  const float* bn2_v  = (const float*)d_in[17];
  const float* gat2_W = (const float*)d_in[18];
  const float* gat2_as= (const float*)d_in[19];
  const float* gat2_ad= (const float*)d_in[20];
  const float* gat2_b = (const float*)d_in[21];

  char* ws = (char*)d_ws;
  size_t off = 0;
  auto alloc = [&](size_t bytes) -> void* {
    void* p = ws + off;
    off += (bytes + 255) & ~(size_t)255;
    return p;
  };
  float*     dinv  = (float*)alloc((size_t)N * 4);
  _Float16*  Xh    = (_Float16*)alloc((size_t)N * 128 * 2);
  _Float16*  Wt    = (_Float16*)alloc((size_t)192 * 128 * 2);
  float*     bufA  = (float*)alloc((size_t)N * 192 * 4);   // GEMM output / gathered h
  float*     bufB  = (float*)alloc((size_t)N * 192 * 4);   // scatter accumulator
  float*     als   = (float*)alloc((size_t)N * 2 * 4);
  float*     ald   = (float*)alloc((size_t)N * 2 * 4);
  unsigned*  emax  = (unsigned*)alloc((size_t)N * 2 * 4);
  float*     denom = (float*)alloc((size_t)N * 2 * 4);

  const int TB = 256;
  auto B1 = [](long n) { return (unsigned)((n + 255) / 256); };       // thread-per-elem
  auto BW = [](long w) { return (unsigned)((w + 7) / 8); };           // 8 waves / block

  // degree (with self loops) -> dinv = rsqrt(deg)
  k_fill_f32<<<B1(N), TB, 0, stream>>>(dinv, 1.0f, N);
  k_deg<<<B1(E), TB, 0, stream>>>(dst, dinv, E);
  k_rsqrt_inplace<<<B1(N), TB, 0, stream>>>(dinv, N);

  // ---- Layer 1: GCN 128 -> 96
  k_cvt_f16<<<B1((long)N * 128), TB, 0, stream>>>(x, Xh, (long)N * 128);
  k_cvt_wT<<<B1(128 * 96), TB, 0, stream>>>(gcn1_W, Wt, 128, 96);
  k_gemm_wmma<<<BW((long)(N / 16) * (96 / 16)), TB, 0, stream>>>(Xh, Wt, bufA, N, 128, 96);
  k_zero_f32<<<B1((long)N * 96), TB, 0, stream>>>(bufB, (long)N * 96);
  k_gcn_scatter<<<BW((long)E + N), TB, 0, stream>>>(bufA, src, dst, dinv, bufB, E, 96);
  k_gcn_post<<<B1((long)N * 96), TB, 0, stream>>>(bufB, gcn1_b, bn1_g, bn1_b, bn1_m, bn1_v,
                                                  Xh, (long)N * 96, 96);

  // ---- Layer 2: GAT 96 -> 2 heads x 96 (mean) + ReLU
  k_cvt_wT<<<B1(96 * 192), TB, 0, stream>>>(gat1_W, Wt, 96, 192);
  k_gemm_wmma<<<BW((long)(N / 16) * (192 / 16)), TB, 0, stream>>>(Xh, Wt, bufA, N, 96, 192);
  k_gat_logits<<<B1((long)N * 2), TB, 0, stream>>>(bufA, gat1_as, gat1_ad, als, ald, N, 2, 96);
  k_zero_f32<<<B1((long)N * 2), TB, 0, stream>>>((float*)emax, (long)N * 2);
  k_att_max<<<B1((long)E + N), TB, 0, stream>>>(src, dst, als, ald, emax, E, 2);
  k_zero_f32<<<B1((long)N * 2), TB, 0, stream>>>(denom, (long)N * 2);
  k_att_denom<<<B1((long)E + N), TB, 0, stream>>>(src, dst, als, ald, emax, denom, E, 2);
  k_zero_f32<<<B1((long)N * 192), TB, 0, stream>>>(bufB, (long)N * 192);
  k_gat_scatter<<<BW((long)E + N), TB, 0, stream>>>(bufA, src, dst, als, ald, emax, denom,
                                                    bufB, E, 2, 96);
  k_gat_post_relu<<<B1((long)N * 96), TB, 0, stream>>>(bufB, gat1_b, Xh, (long)N * 96, 96);

  // ---- Layer 3: GCN 96 -> 96
  k_cvt_wT<<<B1(96 * 96), TB, 0, stream>>>(gcn2_W, Wt, 96, 96);
  k_gemm_wmma<<<BW((long)(N / 16) * (96 / 16)), TB, 0, stream>>>(Xh, Wt, bufA, N, 96, 96);
  k_zero_f32<<<B1((long)N * 96), TB, 0, stream>>>(bufB, (long)N * 96);
  k_gcn_scatter<<<BW((long)E + N), TB, 0, stream>>>(bufA, src, dst, dinv, bufB, E, 96);
  k_gcn_post<<<B1((long)N * 96), TB, 0, stream>>>(bufB, gcn2_b, bn2_g, bn2_b, bn2_m, bn2_v,
                                                  Xh, (long)N * 96, 96);

  // ---- Layer 4: GAT 96 -> 2 heads x 40 (mean), then log_softmax
  k_cvt_wT<<<B1(96 * 80), TB, 0, stream>>>(gat2_W, Wt, 96, 80);
  k_gemm_wmma<<<BW((long)(N / 16) * (80 / 16)), TB, 0, stream>>>(Xh, Wt, bufA, N, 96, 80);
  k_gat_logits<<<B1((long)N * 2), TB, 0, stream>>>(bufA, gat2_as, gat2_ad, als, ald, N, 2, 40);
  k_zero_f32<<<B1((long)N * 2), TB, 0, stream>>>((float*)emax, (long)N * 2);
  k_att_max<<<B1((long)E + N), TB, 0, stream>>>(src, dst, als, ald, emax, E, 2);
  k_zero_f32<<<B1((long)N * 2), TB, 0, stream>>>(denom, (long)N * 2);
  k_att_denom<<<B1((long)E + N), TB, 0, stream>>>(src, dst, als, ald, emax, denom, E, 2);
  k_zero_f32<<<B1((long)N * 80), TB, 0, stream>>>(bufB, (long)N * 80);
  k_gat_scatter<<<BW((long)E + N), TB, 0, stream>>>(bufA, src, dst, als, ald, emax, denom,
                                                    bufB, E, 2, 40);
  k_final_logsoftmax<<<BW(N), TB, 0, stream>>>(bufB, gat2_b, (float*)d_out, N, 40);
}
